// SelfAttention2d_10711648436688
// MI455X (gfx1250) — matrix operand order repaired and checked by reading, not compiled, not run on hardware
//
#include <hip/hip_runtime.h>
#include <hip/hip_bf16.h>

#define BB 8
#define CCH 64
#define NN 4096
#define WAVES 8
#define KT 64     // keys per tile
#define KPAD 72   // halves per lds_k row
#define VPAD 72   // halves per lds_v row
#define PPAD 72   // halves per lds_p row

typedef _Float16 h16;
typedef __attribute__((ext_vector_type(8)))  _Float16 v8h;
typedef __attribute__((ext_vector_type(16))) _Float16 v16h;
typedef __attribute__((ext_vector_type(8)))  float    v8f;

// 16-byte int vector matching the async-LDS builtin's parameter type.
typedef int v4i __attribute__((vector_size(16)));
typedef __attribute__((address_space(1))) v4i gv4i;
typedef __attribute__((address_space(3))) v4i lv4i;

// Build a 16-half fragment from two 16B-aligned 8-half chunks.
__device__ __forceinline__ v16h ld_ab(const h16* p0, const h16* p1) {
  v8h a = *(const v8h*)p0;
  v8h b = *(const v8h*)p1;
  v16h r;
#pragma unroll
  for (int i = 0; i < 8; ++i) { r[i] = a[i]; r[i + 8] = b[i]; }
  return r;
}

__device__ __forceinline__ float fast_exp2(float x) {
#if __has_builtin(__builtin_amdgcn_exp2f)
  return __builtin_amdgcn_exp2f(x);   // v_exp_f32 (native 2^x)
#else
  return exp2f(x);
#endif
}

__device__ __forceinline__ float fast_rcp(float x) {
#if __has_builtin(__builtin_amdgcn_rcpf)
  return __builtin_amdgcn_rcpf(x);    // v_rcp_f32
#else
  return 1.0f / x;
#endif
}

// ---- CDNA5 async global->LDS staging --------------------------------------
#if __has_builtin(__builtin_amdgcn_global_load_async_to_lds_b128)
#define HAVE_ASYNC_LDS 1
#endif

__device__ __forceinline__ void cp_lds16(h16* dst, const h16* src) {
#ifdef HAVE_ASYNC_LDS
  __builtin_amdgcn_global_load_async_to_lds_b128(
      (gv4i*)src, (lv4i*)dst, 0, 0);
#else
  *(v8h*)dst = *(const v8h*)src;
#endif
}

__device__ __forceinline__ void cp_wait() {
#ifdef HAVE_ASYNC_LDS
#if __has_builtin(__builtin_amdgcn_s_wait_asynccnt)
  __builtin_amdgcn_s_wait_asynccnt(0);
#else
  asm volatile("s_wait_asynccnt 0x0" ::: "memory");
#endif
#endif
}

// log2(e) / sqrt(C): folded into Q so softmax runs in base 2.
#define QSCALE 0.1803368801111244f

// ---------------------------------------------------------------------------
// Kernel 1: q/k/v projections. q,k stored (B,N,C) f16; v stored (B,C,N) f16.
// ---------------------------------------------------------------------------
__global__ __launch_bounds__(256) void qkv_proj_kernel(
    const float* __restrict__ x,
    const float* __restrict__ Wq, const float* __restrict__ bq,
    const float* __restrict__ Wk, const float* __restrict__ bk,
    const float* __restrict__ Wv, const float* __restrict__ bv,
    h16* __restrict__ q, h16* __restrict__ k, h16* __restrict__ v)
{
  __shared__ float sW[3][CCH * CCH];
  __shared__ float sB[3][CCH];
  const int t = threadIdx.x;
  for (int i = t; i < CCH * CCH; i += 256) {
    sW[0][i] = Wq[i]; sW[1][i] = Wk[i]; sW[2][i] = Wv[i];
  }
  if (t < CCH) { sB[0][t] = bq[t]; sB[1][t] = bk[t]; sB[2][t] = bv[t]; }
  __syncthreads();

  const int gid = blockIdx.x * 256 + t;      // gid = b*NN + n
  const int b = gid / NN;
  const int n = gid - b * NN;

  float xr[CCH];
  const float* xp = x + (size_t)b * CCH * NN + n;
#pragma unroll
  for (int c = 0; c < CCH; ++c) xr[c] = xp[(size_t)c * NN];

  {  // Q (scaled by log2e/sqrt(C)), layout (B,N,C)
    h16* dst = q + (size_t)gid * CCH;
    for (int d0 = 0; d0 < CCH; d0 += 8) {
      v8h ov;
#pragma unroll
      for (int j = 0; j < 8; ++j) {
        float acc = sB[0][d0 + j];
        const float* w = &sW[0][(d0 + j) * CCH];
#pragma unroll
        for (int c = 0; c < CCH; ++c) acc = fmaf(w[c], xr[c], acc);
        ov[j] = (h16)(acc * QSCALE);
      }
      *(v8h*)(dst + d0) = ov;
    }
  }
  {  // K, layout (B,N,C)
    h16* dst = k + (size_t)gid * CCH;
    for (int d0 = 0; d0 < CCH; d0 += 8) {
      v8h ov;
#pragma unroll
      for (int j = 0; j < 8; ++j) {
        float acc = sB[1][d0 + j];
        const float* w = &sW[1][(d0 + j) * CCH];
#pragma unroll
        for (int c = 0; c < CCH; ++c) acc = fmaf(w[c], xr[c], acc);
        ov[j] = (h16)acc;
      }
      *(v8h*)(dst + d0) = ov;
    }
  }
  {  // V, layout (B,C,N)
    h16* dst = v + (size_t)b * CCH * NN + n;
    for (int d0 = 0; d0 < CCH; d0 += 8) {
#pragma unroll
      for (int j = 0; j < 8; ++j) {
        float acc = sB[2][d0 + j];
        const float* w = &sW[2][(d0 + j) * CCH];
#pragma unroll
        for (int c = 0; c < CCH; ++c) acc = fmaf(w[c], xr[c], acc);
        dst[(size_t)(d0 + j) * NN] = (h16)acc;
      }
    }
  }
}

// ---------------------------------------------------------------------------
// Kernel 2: flash attention with double-buffered async-to-LDS K/V staging.
// One 16-query block per wave; 64-key tiles shared by 8 waves. Row sums via
// WMMA vs all-ones fragment; block-shared base-2 softmax shift.
// ---------------------------------------------------------------------------
__global__ __launch_bounds__(256) void flash_attn_kernel(
    const h16* __restrict__ q, const h16* __restrict__ k,
    const h16* __restrict__ v, float* __restrict__ o)
{
  __shared__ h16 lds_k[2][KT][KPAD];         // 2 x (64 keys x 64 ch)
  __shared__ h16 lds_v[2][CCH][VPAD];        // 2 x (64 ch x 64 keys)
  __shared__ h16 lds_p[WAVES][16][PPAD];     // per-wave P transpose scratch

  const int t    = threadIdx.x;
  const int wid  = t >> 5;
  const int lane = t & 31;
  const int l16  = lane & 15;
  const int lh   = lane >> 4;
  const int b    = blockIdx.y;
  const int n0   = (blockIdx.x * WAVES + wid) * 16;

  const h16* qb = q + (size_t)b * NN * CCH;
  const h16* kb = k + (size_t)b * NN * CCH;
  const h16* vb = v + (size_t)b * CCH * NN;

  // Q A-fragments (16x32 each, channel chunks 0..31 / 32..63)
  v16h qa0, qa1;
  {
    const h16* qrow = qb + (size_t)(n0 + l16) * CCH;
    qa0 = ld_ab(qrow +  0 + lh * 16, qrow +  8 + lh * 16);
    qa1 = ld_ab(qrow + 32 + lh * 16, qrow + 40 + lh * 16);
  }

  // all-ones B fragment for row sums of P
  v16h vones;
#pragma unroll
  for (int i = 0; i < 16; ++i) vones[i] = (h16)1.0f;

  v8f acc0 = {}, acc1 = {}, acc2 = {}, acc3 = {};
  v8f lacc = {};                    // running (rescaled) row sums
  float mrow[8];
#pragma unroll
  for (int r = 0; r < 8; ++r) mrow[r] = -1e30f;

  // cooperative staging: 256 thr x 2 x 8 halves per array
  const int rr0 = (t * 8) >> 6,          cc0 = (t * 8) & 63;
  const int rr1 = (t * 8 + 2048) >> 6,   cc1 = (t * 8 + 2048) & 63;

  // prologue: stage tile 0 into buffer 0
  cp_lds16(&lds_k[0][rr0][cc0], kb + (size_t)rr0 * CCH + cc0);
  cp_lds16(&lds_v[0][rr0][cc0], vb + (size_t)rr0 * NN + cc0);
  cp_lds16(&lds_k[0][rr1][cc1], kb + (size_t)rr1 * CCH + cc1);
  cp_lds16(&lds_v[0][rr1][cc1], vb + (size_t)rr1 * NN + cc1);

  int cur = 0;
  for (int m0 = 0; m0 < NN; m0 += KT) {
    cp_wait();          // my async loads for buffer `cur` have landed
    __syncthreads();    // everyone's loads landed; everyone done reading cur^1

    if (m0 + KT < NN) { // prefetch next tile into the other buffer
      const int nxt = cur ^ 1, m1 = m0 + KT;
      cp_lds16(&lds_k[nxt][rr0][cc0], kb + (size_t)(m1 + rr0) * CCH + cc0);
      cp_lds16(&lds_v[nxt][rr0][cc0], vb + (size_t)rr0 * NN + m1 + cc0);
      cp_lds16(&lds_k[nxt][rr1][cc1], kb + (size_t)(m1 + rr1) * CCH + cc1);
      cp_lds16(&lds_v[nxt][rr1][cc1], vb + (size_t)rr1 * NN + m1 + cc1);
    }

    // ---- scores: 4 C-frags covering 64 key columns ----
    v8f s[4];
#pragma unroll
    for (int mh = 0; mh < 4; ++mh) {
      const h16* kr = &lds_k[cur][mh * 16 + l16][0];
      v16h b0 = ld_ab(kr +  0 + lh * 16, kr +  0 + lh * 16 + 8);
      v16h b1 = ld_ab(kr + 32 + lh * 16, kr + 32 + lh * 16 + 8);
      v8f z = {};
      z     = __builtin_amdgcn_wmma_f32_16x16x32_f16(false, qa0, false, b0, (short)0, z, false, false);
      s[mh] = __builtin_amdgcn_wmma_f32_16x16x32_f16(false, qa1, false, b1, (short)0, z, false, false);
    }

    // ---- block-shared running max (monotone shift; overflow guard only) ----
    float tm = s[0][0];
#pragma unroll
    for (int mh = 0; mh < 4; ++mh)
#pragma unroll
      for (int r = 0; r < 8; ++r) tm = fmaxf(tm, s[mh][r]);
    tm = fmaxf(tm, __shfl_xor(tm, 1));
    tm = fmaxf(tm, __shfl_xor(tm, 2));
    tm = fmaxf(tm, __shfl_xor(tm, 4));
    tm = fmaxf(tm, __shfl_xor(tm, 8));

    // ---- exp2, rescale accumulators, write P to per-wave scratch ----
#pragma unroll
    for (int r = 0; r < 8; ++r) {
      const float mn = fmaxf(mrow[r], tm);
      const float al = fast_exp2(mrow[r] - mn);
      mrow[r] = mn;
      acc0[r] *= al; acc1[r] *= al; acc2[r] *= al; acc3[r] *= al; lacc[r] *= al;
      const int row = r + lh * 8;
#pragma unroll
      for (int mh = 0; mh < 4; ++mh) {
        const float p = fast_exp2(s[mh][r] - mn);
        lds_p[wid][row][mh * 16 + l16] = (h16)p;
      }
    }
    __builtin_amdgcn_wave_barrier();   // wave-local LDS; DScnt keeps order

    // P as two A-fragments (16 rows x keys 0..31 / 32..63)
    const h16* pr = &lds_p[wid][l16][0];
    v16h pa0 = ld_ab(pr +  0 + lh * 16, pr +  8 + lh * 16);
    v16h pa1 = ld_ab(pr + 32 + lh * 16, pr + 40 + lh * 16);

    // row sums via WMMA (broadcast across lanes in the C-frag)
    lacc = __builtin_amdgcn_wmma_f32_16x16x32_f16(false, pa0, false, vones, (short)0, lacc, false, false);
    lacc = __builtin_amdgcn_wmma_f32_16x16x32_f16(false, pa1, false, vones, (short)0, lacc, false, false);

    // ---- O += P * V over 4 output-channel groups of 16 ----
    {
      const h16* vr0 = &lds_v[cur][ 0 + l16][0];
      acc0 = __builtin_amdgcn_wmma_f32_16x16x32_f16(false, pa0, false,
               ld_ab(vr0 + lh * 16, vr0 + lh * 16 + 8), (short)0, acc0, false, false);
      acc0 = __builtin_amdgcn_wmma_f32_16x16x32_f16(false, pa1, false,
               ld_ab(vr0 + 32 + lh * 16, vr0 + 32 + lh * 16 + 8), (short)0, acc0, false, false);
      const h16* vr1 = &lds_v[cur][16 + l16][0];
      acc1 = __builtin_amdgcn_wmma_f32_16x16x32_f16(false, pa0, false,
               ld_ab(vr1 + lh * 16, vr1 + lh * 16 + 8), (short)0, acc1, false, false);
      acc1 = __builtin_amdgcn_wmma_f32_16x16x32_f16(false, pa1, false,
               ld_ab(vr1 + 32 + lh * 16, vr1 + 32 + lh * 16 + 8), (short)0, acc1, false, false);
      const h16* vr2 = &lds_v[cur][32 + l16][0];
      acc2 = __builtin_amdgcn_wmma_f32_16x16x32_f16(false, pa0, false,
               ld_ab(vr2 + lh * 16, vr2 + lh * 16 + 8), (short)0, acc2, false, false);
      acc2 = __builtin_amdgcn_wmma_f32_16x16x32_f16(false, pa1, false,
               ld_ab(vr2 + 32 + lh * 16, vr2 + 32 + lh * 16 + 8), (short)0, acc2, false, false);
      const h16* vr3 = &lds_v[cur][48 + l16][0];
      acc3 = __builtin_amdgcn_wmma_f32_16x16x32_f16(false, pa0, false,
               ld_ab(vr3 + lh * 16, vr3 + lh * 16 + 8), (short)0, acc3, false, false);
      acc3 = __builtin_amdgcn_wmma_f32_16x16x32_f16(false, pa1, false,
               ld_ab(vr3 + 32 + lh * 16, vr3 + 32 + lh * 16 + 8), (short)0, acc3, false, false);
    }
    cur ^= 1;
  }

  float inv[8];
#pragma unroll
  for (int r = 0; r < 8; ++r) inv[r] = fast_rcp(lacc[r]);

  float* ob = o + ((size_t)b * NN + n0) * CCH;
#pragma unroll
  for (int r = 0; r < 8; ++r) {
    const int row = r + lh * 8;
    ob[(size_t)row * CCH +  0 + l16] = acc0[r] * inv[r];
    ob[(size_t)row * CCH + 16 + l16] = acc1[r] * inv[r];
    ob[(size_t)row * CCH + 32 + l16] = acc2[r] * inv[r];
    ob[(size_t)row * CCH + 48 + l16] = acc3[r] * inv[r];
  }
}

// ---------------------------------------------------------------------------
// Kernel 3: output projection + residual, f32.  y = x + Wp @ o + bp
// ---------------------------------------------------------------------------
__global__ __launch_bounds__(256) void out_proj_kernel(
    const float* __restrict__ x, const float* __restrict__ o,
    const float* __restrict__ Wp, const float* __restrict__ bp,
    float* __restrict__ y)
{
  __shared__ float sW[CCH * CCH];
  __shared__ float sB[CCH];
  const int t = threadIdx.x;
  for (int i = t; i < CCH * CCH; i += 256) sW[i] = Wp[i];
  if (t < CCH) sB[t] = bp[t];
  __syncthreads();

  const int gid = blockIdx.x * 256 + t;
  const int b = gid / NN;

  float orow[CCH];
  const float* op = o + (size_t)gid * CCH;
#pragma unroll
  for (int c = 0; c < CCH; ++c) orow[c] = op[c];

  const size_t xbase = (size_t)b * CCH * NN + (gid - b * NN);
  for (int d0 = 0; d0 < CCH; d0 += 8) {
#pragma unroll
    for (int j = 0; j < 8; ++j) {
      const int d = d0 + j;
      float acc = sB[d];
      const float* w = &sW[d * CCH];
#pragma unroll
      for (int c = 0; c < CCH; ++c) acc = fmaf(w[c], orow[c], acc);
      y[xbase + (size_t)d * NN] = x[xbase + (size_t)d * NN] + acc;
    }
  }
}

// ---------------------------------------------------------------------------
extern "C" void kernel_launch(void* const* d_in, const int* in_sizes, int n_in,
                              void* d_out, int out_size, void* d_ws, size_t ws_size,
                              hipStream_t stream) {
  (void)in_sizes; (void)n_in; (void)out_size; (void)ws_size;
  const float* x  = (const float*)d_in[0];
  const float* Wq = (const float*)d_in[1];
  const float* bq = (const float*)d_in[2];
  const float* Wk = (const float*)d_in[3];
  const float* bk = (const float*)d_in[4];
  const float* Wv = (const float*)d_in[5];
  const float* bv = (const float*)d_in[6];
  const float* Wp = (const float*)d_in[7];
  const float* bp = (const float*)d_in[8];

  const size_t elems = (size_t)BB * NN * CCH;     // 2,097,152
  h16*  qws = (h16*)d_ws;                         // 4 MB
  h16*  kws = qws + elems;                        // 4 MB
  h16*  vws = kws + elems;                        // 4 MB
  float* ows = (float*)(vws + elems);             // 8 MB, (B,N,C)

  qkv_proj_kernel<<<(BB * NN) / 256, 256, 0, stream>>>(
      x, Wq, bq, Wk, bk, Wv, bv, qws, kws, vws);

  flash_attn_kernel<<<dim3(NN / (16 * WAVES), BB), 256, 0, stream>>>(
      qws, kws, vws, ows);

  out_proj_kernel<<<(BB * NN) / 256, 256, 0, stream>>>(
      x, ows, Wp, bp, (float*)d_out);
}
